// W4A16NemotronHUniQLMLP_83416854823033
// MI455X (gfx1250) — matrix operand order repaired and checked by reading, not compiled, not run on hardware
//
#include <hip/hip_runtime.h>
#include <hip/hip_bf16.h>

// ---------------------------------------------------------------------------
// W4A16 NemotronH MLP: y = relu(x @ W_up^T)^2 @ W_down^T
// Two WMMA-bf16 GEMMs with fused int4 dequant.
//   - 128x128 block tile, 8 wave32 (2x4), 4x2 WMMA accs/wave
//   - double-buffered LDS (2 x 64KB), one barrier per K-block
//   - GEMM2 A-tiles copied with global_load_async_to_lds_b128 (ASYNCcnt)
//   - packed f32->bf16 conversion (v_cvt_pk_bf16_f32 path)
//   - staging loops partially unrolled to stay under 256 VGPRs
// ---------------------------------------------------------------------------

typedef __attribute__((ext_vector_type(16))) __bf16       v16bf;
typedef __attribute__((ext_vector_type(2)))  __bf16       v2bf;
typedef __attribute__((ext_vector_type(8)))  float        v8f;
typedef __attribute__((ext_vector_type(4)))  unsigned int v4u;

union Frag {
  v4u   q[2];   // 2 x 16B raw chunks
  v16bf bf;     // WMMA operand view
};

// Packed float->bf16 (lets clang select v_cvt_pk_bf16_f32)
__device__ __forceinline__ unsigned int pack2_bf16(float lo, float hi) {
  v2bf p;
  p.x = (__bf16)lo;
  p.y = (__bf16)hi;
  return __builtin_bit_cast(unsigned int, p);
}

// Swizzled element offset inside a 128x128 bf16 LDS tile (16B-chunk XOR
// swizzle: kills fragment-load bank conflicts with zero padding).
__device__ __forceinline__ int lds_off(int row, int col) {
  return row * 128 + (col ^ ((row & 7) << 3));
}

// A_IS_F32:        A operand is fp32 (convert to bf16 while staging via VGPRs),
//                  otherwise A is bf16 and is copied with async global->LDS.
// RELU2_BF16_OUT:  fuse relu(x)^2 and store bf16 (up-proj), else store fp32.
template <bool A_IS_F32, bool RELU2_BF16_OUT>
__global__ void __launch_bounds__(256, 2)
w4a16_wmma_gemm(const void*  __restrict__ Aptr,   // [M,K] fp32 or bf16
                const int*   __restrict__ Q,      // [N,K] int4 in int32
                const float* __restrict__ S,      // [N,K/128]
                const int*   __restrict__ Z,      // [N,K/128]
                void*        __restrict__ Outp,   // [M,N] bf16 or fp32
                int M, int N, int K)
{
  // Dynamic LDS: two buffers of (A tile 16K elems + B tile 16K elems).
  extern __shared__ unsigned short smem[];

  const int tid  = threadIdx.x;
  const int lane = tid & 31;
  const int lr   = lane & 15;   // lane row-in-fragment
  const int lh   = lane >> 4;   // lane half
  const int wave = tid >> 5;    // 8 waves, arranged 2 (M) x 4 (N)
  const int wm0  = (wave >> 2) * 64;
  const int wn0  = (wave & 3)  * 32;

  const long n0 = (long)blockIdx.x * 128;
  const long m0 = (long)blockIdx.y * 128;
  const int  ngrp = K >> 7;     // K/128 quant groups (K-blocks)

  // ---- staging: fill buffer `buf` with K-block `g` ----
  auto stageA = [&](int buf, int g) {
    const long k0 = (long)g << 7;
    if constexpr (A_IS_F32) {
      unsigned short* As = smem + buf * 32768;
      const float* A = (const float*)Aptr;
      // Partial unroll: 8 float4 in flight (32 VGPRs) keeps us < 256 VGPRs.
      #pragma unroll 8
      for (int i = 0; i < 16; ++i) {
        const int slot = tid + (i << 8);
        const int r  = slot >> 5;
        const int c4 = slot & 31;
        const float4 v = *(const float4*)(A + (m0 + r) * (long)K + k0 + c4 * 4);
        uint2 pk;
        pk.x = pack2_bf16(v.x, v.y);
        pk.y = pack2_bf16(v.z, v.w);
        *(uint2*)&As[lds_off(r, c4 * 4)] = pk;
      }
    } else {
      // Async copy: 2048 16B chunks, 8 per thread; no VGPR data staging, so
      // these overlap with the WMMA stream of the previous K-block.
      const unsigned short* A = (const unsigned short*)Aptr;
      #pragma unroll
      for (int i = 0; i < 8; ++i) {
        const int cid = tid + (i << 8);
        const int r   = cid >> 4;
        const int c8  = (cid & 15) << 3;          // element col, 8-aligned
        const unsigned ldsb =
            (unsigned)((buf * 32768 + lds_off(r, c8)) * 2);  // byte offset
        const unsigned short* gp = A + (m0 + r) * (long)K + k0 + c8;
        asm volatile("global_load_async_to_lds_b128 %0, %1, off"
                     :: "v"(ldsb), "v"(gp) : "memory");
      }
    }
  };

  auto stageB = [&](int buf, int g) {
    unsigned short* Bs = smem + buf * 32768 + 16384;
    const long k0 = (long)g << 7;
    // Partial unroll: 8 int4 in flight (32 VGPRs).
    #pragma unroll 8
    for (int i = 0; i < 16; ++i) {
      const int slot = tid + (i << 8);
      const int r  = slot >> 5;
      const int c4 = slot & 31;
      const long nrow = n0 + r;
      const int4  q  = *(const int4*)(Q + nrow * (long)K + k0 + c4 * 4);
      const float s  = S[nrow * (long)ngrp + g];
      const float zs = -(float)Z[nrow * (long)ngrp + g] * s;  // w = q*s + zs
      const float w0 = fmaf((float)q.x, s, zs);
      const float w1 = fmaf((float)q.y, s, zs);
      const float w2 = fmaf((float)q.z, s, zs);
      const float w3 = fmaf((float)q.w, s, zs);
      uint2 pk;
      pk.x = pack2_bf16(w0, w1);
      pk.y = pack2_bf16(w2, w3);
      *(uint2*)&Bs[lds_off(r, c4 * 4)] = pk;
    }
  };

  v8f acc[4][2];
  #pragma unroll
  for (int mi = 0; mi < 4; ++mi)
    #pragma unroll
    for (int ni = 0; ni < 2; ++ni)
      acc[mi][ni] = (v8f){};

  // ---- software pipeline: stage(g+1) overlaps WMMA(g) ----
  stageA(0, 0);
  stageB(0, 0);

  for (int g = 0; g < ngrp; ++g) {
    const int buf = g & 1;
    if constexpr (!A_IS_F32) {
      // drain async fills of this iteration's buffer before anyone reads it
      asm volatile("s_wait_asynccnt 0" ::: "memory");
    }
    __syncthreads();

    if (g + 1 < ngrp) {
      stageA(buf ^ 1, g + 1);
      stageB(buf ^ 1, g + 1);
    }

    const unsigned short* As = smem + buf * 32768;
    const unsigned short* Bs = As + 16384;

    #pragma unroll
    for (int kk = 0; kk < 4; ++kk) {
      Frag a[4], b[2];
      // A 16x32 bf16 fragment (ISA 7.12.2): lane half h holds
      // K = {8h..8h+7} and {16+8h..16+8h+7} -> two 16B chunks.
      #pragma unroll
      for (int mi = 0; mi < 4; ++mi) {
        const int row = wm0 + mi * 16 + lr;
        a[mi].q[0] = *(const v4u*)&As[lds_off(row, kk * 32 + lh * 8)];
        a[mi].q[1] = *(const v4u*)&As[lds_off(row, kk * 32 + lh * 8 + 16)];
      }
      // B 32x16 bf16 fragment: lane half h holds K = {8h..8h+7} and {16+8h..16+8h+7}.
      #pragma unroll
      for (int ni = 0; ni < 2; ++ni) {
        const int row = wn0 + ni * 16 + lr;   // row in Bs == output column n
        b[ni].q[0] = *(const v4u*)&Bs[lds_off(row, kk * 32 + lh * 8)];
        b[ni].q[1] = *(const v4u*)&Bs[lds_off(row, kk * 32 + lh * 8 + 16)];
      }
      #pragma unroll
      for (int mi = 0; mi < 4; ++mi)
        #pragma unroll
        for (int ni = 0; ni < 2; ++ni)
          acc[mi][ni] = __builtin_amdgcn_wmma_f32_16x16x32_bf16(
              false, a[mi].bf, false, b[ni].bf,
              (short)0, acc[mi][ni], false, false);
    }
  }

  // ---- epilogue: C/D layout = (row: vgpr + 8*lh, col: lr) ----
  #pragma unroll
  for (int mi = 0; mi < 4; ++mi) {
    #pragma unroll
    for (int ni = 0; ni < 2; ++ni) {
      const long col = n0 + wn0 + ni * 16 + lr;
      #pragma unroll
      for (int r = 0; r < 8; ++r) {
        const long row = m0 + wm0 + mi * 16 + lh * 8 + r;
        float v = acc[mi][ni][r];
        if constexpr (RELU2_BF16_OUT) {
          v = v > 0.0f ? v * v : 0.0f;   // relu(x)^2
          ((unsigned short*)Outp)[row * (long)N + col] =
              __builtin_bit_cast(unsigned short, (__bf16)v);
        } else {
          ((float*)Outp)[row * (long)N + col] = v;
        }
      }
    }
  }
}

extern "C" void kernel_launch(void* const* d_in, const int* in_sizes, int n_in,
                              void* d_out, int out_size, void* d_ws, size_t ws_size,
                              hipStream_t stream) {
  const float* x      = (const float*)d_in[0];  // [2,2048,H] fp32
  const int*   q_up   = (const int*)d_in[1];    // [I,H]
  const float* s_up   = (const float*)d_in[2];  // [I,H/128]
  const int*   z_up   = (const int*)d_in[3];    // [I,H/128]
  const int*   q_down = (const int*)d_in[4];    // [H,I]
  const float* s_down = (const float*)d_in[5];  // [H,I/128]
  const int*   z_down = (const int*)d_in[6];    // [H,I/128]
  float*          out = (float*)d_out;          // [2,2048,H] fp32
  unsigned short* act = (unsigned short*)d_ws;  // bf16 [M,I] intermediate

  // Derive dims: |q_up| = I*H, |s_up| = I*H/128, |x| = M*H
  const long H = 128L * (long)in_sizes[1] / (long)in_sizes[2];  // 4096
  const long I = (long)in_sizes[1] / H;                         // 14336
  const long Mtok = (long)in_sizes[0] / H;                      // 4096

  dim3 blk(256, 1, 1);
  const size_t lds_bytes = 2 * 2 * 128 * 128 * sizeof(unsigned short); // 128KB

  // GEMM1: act = relu(x @ W_up^T)^2, bf16 out -> d_ws
  dim3 g1((unsigned)(I / 128), (unsigned)(Mtok / 128), 1);
  hipLaunchKernelGGL((w4a16_wmma_gemm<true, true>), g1, blk, lds_bytes, stream,
                     (const void*)x, q_up, s_up, z_up, (void*)act,
                     (int)Mtok, (int)I, (int)H);

  // GEMM2: out = act @ W_down^T, fp32 out -> d_out
  dim3 g2((unsigned)(H / 128), (unsigned)(Mtok / 128), 1);
  hipLaunchKernelGGL((w4a16_wmma_gemm<false, false>), g2, blk, lds_bytes, stream,
                     (const void*)act, q_down, s_down, z_down, (void*)out,
                     (int)Mtok, (int)H, (int)I);
}
